// CrossAttention_12584254177448
// MI455X (gfx1250) — matrix operand order repaired and checked by reading, not compiled, not run on hardware
//
#include <hip/hip_runtime.h>
#include <hip/hip_bf16.h>

// ---------------------------------------------------------------------------
// CrossAttention for MI455X (gfx1250): bf16 WMMA GEMMs, double-buffered LDS,
// software-pipelined global loads, masked softmax.
//   query:[32,1024,1024] f32, key/value same, mask:[32,1024] i32,
//   W_w:[1024,1024] f32, W_b:[1024] f32.
//   out = attended [32,1024,1024] f32 ++ attention_weights [32,1024,1024] f32
// ---------------------------------------------------------------------------

typedef __bf16 bf16_t;
typedef bf16_t v16bf __attribute__((ext_vector_type(16)));
typedef float  v8f   __attribute__((ext_vector_type(8)));

#define EMBED 1024
#define BATCH 32
#define QLEN  1024
#define KLEN  1024
#define SCALE 0.03125f  // 1/sqrt(1024)

// GEMM tiling: 256 threads = 8 waves (4x2); workgroup tile 128x128;
// wave tile 32x64 = 2x4 WMMA tiles of 16x16; K staged in chunks of 32.
#define BM 128
#define BN 128
#define BK 32
#define LSTR 40  // padded LDS row stride (bf16) to dodge bank conflicts

// ---- staging helpers ------------------------------------------------------
__device__ __forceinline__ void ldg4(const float* __restrict__ p, float4 r[4]) {
  const float4* s = (const float4*)p;
#pragma unroll
  for (int i = 0; i < 4; ++i) r[i] = s[i];
}
__device__ __forceinline__ void st_bf16x16(bf16_t* __restrict__ dst, const float4 r[4]) {
#pragma unroll
  for (int i = 0; i < 4; ++i) {
    dst[4*i+0] = (bf16_t)r[i].x; dst[4*i+1] = (bf16_t)r[i].y;
    dst[4*i+2] = (bf16_t)r[i].z; dst[4*i+3] = (bf16_t)r[i].w;
  }
}

// A fragment (16x32 bf16, wave32): lanes 0-15 hold M=lane; VGPR p<4 -> K=2p..,
// VGPR p>=4 -> K=16+..; lanes 16-31 shifted by 8 in K (ISA 7.12.2).
__device__ __forceinline__ v16bf load_a_frag(const bf16_t* __restrict__ row, int half) {
  v16bf a;
#pragma unroll
  for (int p = 0; p < 8; ++p) {
    int kb = ((p >= 4) ? 16 : 0) + (half ? 8 : 0) + 2 * (p & 3);
    a[2 * p]     = row[kb];
    a[2 * p + 1] = row[kb + 1];
  }
  return a;
}
// B fragment (32x16 bf16): lane holds column N=lane%16; same K order as A:
// VGPR p<4 -> K=2p.., VGPR p>=4 -> K=16+..; lanes 16-31 shifted by 8 in K.
__device__ __forceinline__ v16bf load_b_frag(const bf16_t* __restrict__ col, int half) {
  v16bf b;
#pragma unroll
  for (int p = 0; p < 8; ++p) {
    int kb = ((p >= 4) ? 16 : 0) + (half ? 8 : 0) + 2 * (p & 3);
    b[2 * p]     = col[kb];
    b[2 * p + 1] = col[kb + 1];
  }
  return b;
}

// 8 WMMAs for one K-chunk: wave tile 32x64 from staged LDS buffers.
__device__ __forceinline__ void wave_mma(const bf16_t* __restrict__ Abase,
                                         const bf16_t* __restrict__ Bbase,
                                         int waveM, int waveN, int l16, int half,
                                         v8f acc[2][4]) {
  v16bf afrag[2];
#pragma unroll
  for (int tm = 0; tm < 2; ++tm)
    afrag[tm] = load_a_frag(Abase + (waveM * 32 + tm * 16 + l16) * LSTR, half);
#pragma unroll
  for (int tn = 0; tn < 4; ++tn) {
    v16bf bfrag = load_b_frag(Bbase + (waveN * 64 + tn * 16 + l16) * LSTR, half);
#pragma unroll
    for (int tm = 0; tm < 2; ++tm)
      acc[tm][tn] = __builtin_amdgcn_wmma_f32_16x16x32_bf16(
          false, afrag[tm], false, bfrag, (short)0, acc[tm][tn], false, false);
  }
}

// ---------------------------------------------------------------------------
// Kernel 1: query_proj = query @ W^T + bias -> bf16 workspace.
// ---------------------------------------------------------------------------
__global__ __launch_bounds__(256)
void proj_kernel(const float* __restrict__ Q, const float* __restrict__ W,
                 const float* __restrict__ bias, bf16_t* __restrict__ QP) {
  __shared__ bf16_t As[2][BM * LSTR];
  __shared__ bf16_t Bs[2][BN * LSTR];
  const int t = threadIdx.x;
  const int lane = t & 31, wave = t >> 5;
  const int waveM = wave >> 1, waveN = wave & 1;
  const int half = lane >> 4, l16 = lane & 15;
  const int m0 = blockIdx.y * BM;
  const int n0 = blockIdx.x * BN;

  const int sr = t >> 1, sc = (t & 1) * 16;  // staging: 128 rows x 32 cols
  const float* aSrc = Q + (size_t)(m0 + sr) * EMBED + sc;
  const float* bSrc = W + (size_t)(n0 + sr) * EMBED + sc;
  bf16_t* aDst = &As[0][0] + sr * LSTR + sc;
  bf16_t* bDst = &Bs[0][0] + sr * LSTR + sc;
  const int bufStrideA = BM * LSTR, bufStrideB = BN * LSTR;

  v8f acc[2][4] = {};
  float4 aReg[4], bReg[4];

  ldg4(aSrc, aReg); ldg4(bSrc, bReg);
  st_bf16x16(aDst, aReg); st_bf16x16(bDst, bReg);
  __syncthreads();

  const int NCH = EMBED / BK;
  for (int i = 0; i < NCH; ++i) {
    const int cur = i & 1;
    const bool more = (i + 1) < NCH;
    if (more) { ldg4(aSrc + (i + 1) * BK, aReg); ldg4(bSrc + (i + 1) * BK, bReg); }
    wave_mma(&As[0][0] + cur * bufStrideA, &Bs[0][0] + cur * bufStrideB,
             waveM, waveN, l16, half, acc);
    if (more) {
      st_bf16x16(aDst + (1 - cur) * bufStrideA, aReg);
      st_bf16x16(bDst + (1 - cur) * bufStrideB, bReg);
    }
    __syncthreads();
  }

#pragma unroll
  for (int tm = 0; tm < 2; ++tm)
#pragma unroll
    for (int tn = 0; tn < 4; ++tn) {
      int col = n0 + waveN * 64 + tn * 16 + l16;
      float bv = bias[col];
#pragma unroll
      for (int v = 0; v < 8; ++v) {
        int row = m0 + waveM * 32 + tm * 16 + v + half * 8;  // C/D layout 7.12.2
        QP[(size_t)row * EMBED + col] = (bf16_t)(acc[tm][tn][v] + bv);
      }
    }
}

// ---------------------------------------------------------------------------
// Kernel 2: scores = (qp @ key^T) * SCALE per batch -> f32 attn region.
// ---------------------------------------------------------------------------
__global__ __launch_bounds__(256)
void scores_kernel(const bf16_t* __restrict__ QP, const float* __restrict__ Key,
                   float* __restrict__ Attn) {
  __shared__ bf16_t As[2][BM * LSTR];
  __shared__ bf16_t Bs[2][BN * LSTR];
  const int t = threadIdx.x;
  const int lane = t & 31, wave = t >> 5;
  const int waveM = wave >> 1, waveN = wave & 1;
  const int half = lane >> 4, l16 = lane & 15;
  const int b = blockIdx.z;
  const int m0 = blockIdx.y * BM;
  const int n0 = blockIdx.x * BN;

  const bf16_t* Ab = QP + (size_t)b * QLEN * EMBED;
  const float*  Bb = Key + (size_t)b * KLEN * EMBED;
  float*        Ob = Attn + (size_t)b * QLEN * KLEN;

  const int sr = t >> 1, sc = (t & 1) * 16;
  const bf16_t* aSrc = Ab + (size_t)(m0 + sr) * EMBED + sc;  // already bf16
  const float*  bSrc = Bb + (size_t)(n0 + sr) * EMBED + sc;
  bf16_t* aDst = &As[0][0] + sr * LSTR + sc;
  bf16_t* bDst = &Bs[0][0] + sr * LSTR + sc;
  const int bufStrideA = BM * LSTR, bufStrideB = BN * LSTR;

  v8f acc[2][4] = {};
  uint4 aReg[2];
  float4 bReg[4];

  { const uint4* s = (const uint4*)aSrc; aReg[0] = s[0]; aReg[1] = s[1]; }
  ldg4(bSrc, bReg);
  { uint4* d = (uint4*)aDst; d[0] = aReg[0]; d[1] = aReg[1]; }
  st_bf16x16(bDst, bReg);
  __syncthreads();

  const int NCH = EMBED / BK;
  for (int i = 0; i < NCH; ++i) {
    const int cur = i & 1;
    const bool more = (i + 1) < NCH;
    if (more) {
      const uint4* s = (const uint4*)(aSrc + (i + 1) * BK);
      aReg[0] = s[0]; aReg[1] = s[1];
      ldg4(bSrc + (i + 1) * BK, bReg);
    }
    wave_mma(&As[0][0] + cur * bufStrideA, &Bs[0][0] + cur * bufStrideB,
             waveM, waveN, l16, half, acc);
    if (more) {
      uint4* d = (uint4*)(aDst + (1 - cur) * bufStrideA);
      d[0] = aReg[0]; d[1] = aReg[1];
      st_bf16x16(bDst + (1 - cur) * bufStrideB, bReg);
    }
    __syncthreads();
  }

#pragma unroll
  for (int tm = 0; tm < 2; ++tm)
#pragma unroll
    for (int tn = 0; tn < 4; ++tn) {
      int col = n0 + waveN * 64 + tn * 16 + l16;
#pragma unroll
      for (int v = 0; v < 8; ++v) {
        int row = m0 + waveM * 32 + tm * 16 + v + half * 8;
        Ob[(size_t)row * KLEN + col] = acc[tm][tn][v] * SCALE;
      }
    }
}

// ---------------------------------------------------------------------------
// Kernel 3: masked row softmax in-place, one 256-thread block per (b,q) row.
// ---------------------------------------------------------------------------
__global__ __launch_bounds__(256)
void softmax_kernel(float* __restrict__ Attn, const int* __restrict__ mask) {
  const int row = blockIdx.x;     // 0 .. B*Q-1
  const int b = row >> 10;        // row / QLEN
  float* p = Attn + (size_t)row * KLEN;
  const int* mk = mask + (size_t)b * KLEN;

  __shared__ float red[8];
  __shared__ float bcast;
  const int t = threadIdx.x;
  const int lane = t & 31, wave = t >> 5;

  float vals[4];
  float mx = -INFINITY;
#pragma unroll
  for (int i = 0; i < 4; ++i) {
    int k = t + i * 256;
    float s = p[k];
    s = (mk[k] == 0) ? -INFINITY : s;
    vals[i] = s;
    mx = fmaxf(mx, s);
  }
#pragma unroll
  for (int off = 16; off; off >>= 1) mx = fmaxf(mx, __shfl_xor(mx, off, 32));
  if (lane == 0) red[wave] = mx;
  __syncthreads();
  if (t == 0) {
    float m = red[0];
#pragma unroll
    for (int i = 1; i < 8; ++i) m = fmaxf(m, red[i]);
    bcast = m;
  }
  __syncthreads();
  const float m = bcast;

  float sum = 0.f;
#pragma unroll
  for (int i = 0; i < 4; ++i) {
    vals[i] = __expf(vals[i] - m);
    sum += vals[i];
  }
#pragma unroll
  for (int off = 16; off; off >>= 1) sum += __shfl_xor(sum, off, 32);
  __syncthreads();
  if (lane == 0) red[wave] = sum;
  __syncthreads();
  if (t == 0) {
    float s = 0.f;
#pragma unroll
    for (int i = 0; i < 8; ++i) s += red[i];
    bcast = s;
  }
  __syncthreads();
  const float inv = 1.0f / bcast;
#pragma unroll
  for (int i = 0; i < 4; ++i) p[t + i * 256] = vals[i] * inv;
}

// ---------------------------------------------------------------------------
// Kernel 4: attended = attn @ value per batch.  B col n -> value column d=n
// (strided k-gather, coalesced along n).
// ---------------------------------------------------------------------------
__global__ __launch_bounds__(256)
void av_kernel(const float* __restrict__ Attn, const float* __restrict__ Val,
               float* __restrict__ Out) {
  __shared__ bf16_t As[2][BM * LSTR];
  __shared__ bf16_t Bs[2][BN * LSTR];
  const int t = threadIdx.x;
  const int lane = t & 31, wave = t >> 5;
  const int waveM = wave >> 1, waveN = wave & 1;
  const int half = lane >> 4, l16 = lane & 15;
  const int b = blockIdx.z;
  const int m0 = blockIdx.y * BM;
  const int n0 = blockIdx.x * BN;

  const float* Ab = Attn + (size_t)b * QLEN * KLEN;
  const float* Bb = Val + (size_t)b * KLEN * EMBED;
  float*       Ob = Out + (size_t)b * QLEN * EMBED;

  const int sr = t >> 1, sc = (t & 1) * 16;       // A staging: 128 x 32
  const int bn = t & 127, bk = (t >> 7) * 16;     // B staging: 128 n x 32 k
  const float* aSrc = Ab + (size_t)(m0 + sr) * KLEN + sc;
  const float* bSrc = Bb + n0 + bn;               // + (k0+bk+j)*EMBED
  bf16_t* aDst = &As[0][0] + sr * LSTR + sc;
  bf16_t* bDst = &Bs[0][0] + bn * LSTR + bk;
  const int bufStrideA = BM * LSTR, bufStrideB = BN * LSTR;

  v8f acc[2][4] = {};
  float4 aReg[4];
  float  bReg[16];

  ldg4(aSrc, aReg);
#pragma unroll
  for (int j = 0; j < 16; ++j) bReg[j] = bSrc[(size_t)(bk + j) * EMBED];
  st_bf16x16(aDst, aReg);
#pragma unroll
  for (int j = 0; j < 16; ++j) bDst[j] = (bf16_t)bReg[j];
  __syncthreads();

  const int NCH = KLEN / BK;
  for (int i = 0; i < NCH; ++i) {
    const int cur = i & 1;
    const bool more = (i + 1) < NCH;
    if (more) {
      ldg4(aSrc + (i + 1) * BK, aReg);
#pragma unroll
      for (int j = 0; j < 16; ++j)
        bReg[j] = bSrc[(size_t)((i + 1) * BK + bk + j) * EMBED];
    }
    wave_mma(&As[0][0] + cur * bufStrideA, &Bs[0][0] + cur * bufStrideB,
             waveM, waveN, l16, half, acc);
    if (more) {
      st_bf16x16(aDst + (1 - cur) * bufStrideA, aReg);
      bf16_t* d = bDst + (1 - cur) * bufStrideB;
#pragma unroll
      for (int j = 0; j < 16; ++j) d[j] = (bf16_t)bReg[j];
    }
    __syncthreads();
  }

#pragma unroll
  for (int tm = 0; tm < 2; ++tm)
#pragma unroll
    for (int tn = 0; tn < 4; ++tn) {
      int col = n0 + waveN * 64 + tn * 16 + l16;
#pragma unroll
      for (int v = 0; v < 8; ++v) {
        int row = m0 + waveM * 32 + tm * 16 + v + half * 8;
        Ob[(size_t)row * EMBED + col] = acc[tm][tn][v];
      }
    }
}

// ---------------------------------------------------------------------------
extern "C" void kernel_launch(void* const* d_in, const int* in_sizes, int n_in,
                              void* d_out, int out_size, void* d_ws, size_t ws_size,
                              hipStream_t stream) {
  (void)in_sizes; (void)n_in; (void)out_size; (void)ws_size;
  const float* query = (const float*)d_in[0];
  const float* key   = (const float*)d_in[1];
  const float* value = (const float*)d_in[2];
  const int*   mask  = (const int*)d_in[3];
  const float* W_w   = (const float*)d_in[4];
  const float* W_b   = (const float*)d_in[5];

  float* out      = (float*)d_out;
  float* attended = out;                                  // [B,Q,D]
  float* attn     = out + (size_t)BATCH * QLEN * EMBED;   // [B,Q,K]
  bf16_t* qp      = (bf16_t*)d_ws;                        // [B*Q, D] bf16

  {  // 1) projection: M = B*Q = 32768, N = D = 1024
    dim3 grid(EMBED / BN, (BATCH * QLEN) / BM);
    proj_kernel<<<grid, 256, 0, stream>>>(query, W_w, W_b, qp);
  }
  {  // 2) scores -> attn region (pre-softmax)
    dim3 grid(KLEN / BN, QLEN / BM, BATCH);
    scores_kernel<<<grid, 256, 0, stream>>>(qp, key, attn);
  }
  {  // 3) masked softmax in place
    dim3 grid(BATCH * QLEN);
    softmax_kernel<<<grid, 256, 0, stream>>>(attn, mask);
  }
  {  // 4) attended = attn @ value
    dim3 grid(EMBED / BN, QLEN / BM, BATCH);
    av_kernel<<<grid, 256, 0, stream>>>(attn, value, attended);
  }
}
